// RecurrentMusicGenerator_88562225643808
// MI455X (gfx1250) — matrix operand order repaired and checked by reading, not compiled, not run on hardware
//
#include <hip/hip_runtime.h>
#include <hip/hip_bf16.h>

// ---------------------------------------------------------------------------
// RecurrentMusicGenerator on MI455X (gfx1250, wave32, WMMA)
//   L=6, D=512, H=8, E=64, F=2048, N=512, MIX*3=768
// Strategy:
//   * all GEMMs via v_wmma_f32_16x16x32_f16 (f16 in, f32 accumulate)
//   * weights converted fp32 -> f16-transposed on the fly (reused 2MB slab)
//   * attention S-state update = streaming kernel (bandwidth bound, ~805MB)
// ---------------------------------------------------------------------------

typedef _Float16 vh8  __attribute__((ext_vector_type(8)));
typedef _Float16 vh16 __attribute__((ext_vector_type(16)));
typedef float    vf8  __attribute__((ext_vector_type(8)));

union Pack16 { vh16 v; vh8 h[2]; };

#define NLAYER 6
#define DMODEL 512
#define FDIM   2048

// ---------------- weight convert + transpose: W[K][N] f32 -> Wt[N][K] f16 ---
__global__ __launch_bounds__(256)
void k_convert_wt(const float* __restrict__ W, _Float16* __restrict__ Wt,
                  int K, int Nout) {
  int idx = blockIdx.x * 256 + threadIdx.x;
  if (idx >= K * Nout) return;
  int k = idx / Nout, n = idx - k * Nout;
  Wt[(size_t)n * K + k] = (_Float16)W[idx];
}

// ---------------- embedding + positional concat -----------------------------
__global__ __launch_bounds__(256)
void k_embed(const int* __restrict__ x, const int* __restrict__ ip,
             const float* __restrict__ emb, const float* __restrict__ pe,
             float* __restrict__ h, _Float16* __restrict__ h16) {
  int idx = blockIdx.x * 256 + threadIdx.x;   // over N*D = 512*512
  int n = idx >> 9, d = idx & 511;
  float v = (d < 256) ? emb[x[n] * 256 + d] : pe[ip[0] * 256 + (d - 256)];
  h[idx]   = v;
  h16[idx] = (_Float16)v;
}

// ---------------- WMMA GEMM: C = act(A[MxK] * Bt[NxK]^T + bias) (+res) ------
// act: 0=none, 1=phi (elu+1), 2=relu
__global__ __launch_bounds__(256)
void k_gemm_wmma(const _Float16* __restrict__ A, const _Float16* __restrict__ Bt,
                 const float* __restrict__ bias, const float* __restrict__ residual,
                 float* __restrict__ C, _Float16* __restrict__ C16,
                 int M, int K, int Nout, int act) {
  const int lane = threadIdx.x & 31;
  const int wave = threadIdx.x >> 5;        // 8 waves: 4 (M) x 2 (N)
  const int l15  = lane & 15;
  const int g    = lane >> 4;               // half-wave group
  const int mrow = blockIdx.x * 64 + (wave & 3) * 16;
  const int ncol = blockIdx.y * 64 + (wave >> 2) * 32;

  // A fragment rows / B fragment cols per ISA 7.12.2 layouts
  const _Float16* Arow  = A  + (size_t)(mrow + l15) * K;
  const _Float16* Brow0 = Bt + (size_t)(ncol + l15) * K;
  const _Float16* Brow1 = Brow0 + (size_t)16 * K;

  vf8 acc0 = {};
  vf8 acc1 = {};
  for (int k0 = 0; k0 < K; k0 += 32) {
    Pack16 a, b0, b1;
    // A 16x32 f16: lanes 0-15 K={0..7,16..23}, lanes 16-31 K={8..15,24..31}
    a.h[0]  = *(const vh8*)(Arow  + k0 + g * 8);
    a.h[1]  = *(const vh8*)(Arow  + k0 + 16 + g * 8);
    // B 32x16 f16 (stored transposed): lanes 0-15 K={0..7,16..23}, lanes 16-31 K={8..15,24..31}
    b0.h[0] = *(const vh8*)(Brow0 + k0 + g * 8);
    b0.h[1] = *(const vh8*)(Brow0 + k0 + 16 + g * 8);
    b1.h[0] = *(const vh8*)(Brow1 + k0 + g * 8);
    b1.h[1] = *(const vh8*)(Brow1 + k0 + 16 + g * 8);
    acc0 = __builtin_amdgcn_wmma_f32_16x16x32_f16(false, a.v, false, b0.v,
                                                  (short)0, acc0, false, false);
    acc1 = __builtin_amdgcn_wmma_f32_16x16x32_f16(false, a.v, false, b1.v,
                                                  (short)0, acc1, false, false);
  }

  // C/D layout: lane l: n = l&15; VGPR i: m = i + 8*(l>>4)
  const int n0 = ncol + l15;
  const int n1 = n0 + 16;
#pragma unroll
  for (int i2 = 0; i2 < 8; ++i2) {
    int m = mrow + i2 + 8 * g;
    float u0 = acc0[i2], u1 = acc1[i2];
    if (bias) { u0 += bias[n0]; u1 += bias[n1]; }
    if (act == 1) {              // phi = elu + 1
      u0 = (u0 > 0.f) ? (u0 + 1.f) : __expf(u0);
      u1 = (u1 > 0.f) ? (u1 + 1.f) : __expf(u1);
    } else if (act == 2) {       // relu
      u0 = fmaxf(u0, 0.f);
      u1 = fmaxf(u1, 0.f);
    }
    size_t o0 = (size_t)m * Nout + n0;
    size_t o1 = o0 + 16;
    if (residual) { u0 += residual[o0]; u1 += residual[o1]; }
    if (C)   { C[o0] = u0;              C[o1] = u1; }
    if (C16) { C16[o0] = (_Float16)u0;  C16[o1] = (_Float16)u1; }
  }
}

// ---------------- linear-attention state update + readout -------------------
// One block per (n,h). S = Si + K outer V ; Z = Zi + K ; attn = Q@S / (Q.Z+eps)
__global__ __launch_bounds__(256)
void k_attn_state(const float* __restrict__ Q, const float* __restrict__ Kf,
                  const float* __restrict__ V, const float* __restrict__ Si,
                  const float* __restrict__ Zi, float* __restrict__ Sout,
                  float* __restrict__ Zout, _Float16* __restrict__ attn16) {
  __shared__ float q[64], kk[64], vv[64], zz[64], accs[4][64];
  const int nh = blockIdx.x;               // n*H + h
  const int t  = threadIdx.x;
  const size_t vb = (size_t)nh * 64;       // == n*512 + h*64 into (N, H*E)
  if (t < 64) {
    float kv = Kf[vb + t];
    q[t]  = Q[vb + t];
    kk[t] = kv;
    vv[t] = V[vb + t];
    float z = Zi[vb + t] + kv;
    zz[t] = z;
    Zout[vb + t] = z;
  }
  __syncthreads();
  const int m  = t & 63;                   // output column (coalesced streams)
  const int eb = (t >> 6) << 4;            // 16-row strip of e
  const size_t sb = (size_t)nh * 4096;
  const float vm = vv[m];
  float acc = 0.f;
#pragma unroll
  for (int j = 0; j < 16; ++j) {
    int e = eb + j;
    size_t off = sb + ((size_t)e << 6) + m;
    __builtin_prefetch(Si + off + 2048, 0, 0);     // gfx1250 global_prefetch
    float s = Si[off] + kk[e] * vm;
    Sout[off] = s;
    acc = fmaf(q[e], s, acc);
  }
  accs[t >> 6][m] = acc;
  __syncthreads();
  if (t < 64) {
    float den = 1e-6f;
#pragma unroll 8
    for (int e = 0; e < 64; ++e) den = fmaf(q[e], zz[e], den);
    float a = (accs[0][t] + accs[1][t] + accs[2][t] + accs[3][t]) / den;
    attn16[vb + t] = (_Float16)a;
  }
}

// ---------------- layernorm (optional residual), one block per row ----------
__global__ __launch_bounds__(256)
void k_layernorm(const float* __restrict__ x, const float* __restrict__ res,
                 const float* __restrict__ g, const float* __restrict__ b,
                 float* __restrict__ out, _Float16* __restrict__ out16) {
  __shared__ float red[256];
  const int row = blockIdx.x, t = threadIdx.x;
  const size_t base = (size_t)row * 512;
  float v0 = x[base + t], v1 = x[base + t + 256];
  if (res) { v0 += res[base + t]; v1 += res[base + t + 256]; }
  red[t] = v0 + v1;
  __syncthreads();
  for (int s = 128; s > 0; s >>= 1) { if (t < s) red[t] += red[t + s]; __syncthreads(); }
  float mean = red[0] * (1.f / 512.f);
  __syncthreads();
  float d0 = v0 - mean, d1 = v1 - mean;
  red[t] = d0 * d0 + d1 * d1;
  __syncthreads();
  for (int s = 128; s > 0; s >>= 1) { if (t < s) red[t] += red[t + s]; __syncthreads(); }
  float rstd = rsqrtf(red[0] * (1.f / 512.f) + 1e-5f);
  float o0 = d0 * rstd * g[t] + b[t];
  float o1 = d1 * rstd * g[t + 256] + b[t + 256];
  out[base + t]       = o0;
  out[base + t + 256] = o1;
  if (out16) {
    out16[base + t]       = (_Float16)o0;
    out16[base + t + 256] = (_Float16)o1;
  }
}

// ---------------------------------------------------------------------------
extern "C" void kernel_launch(void* const* d_in, const int* in_sizes, int n_in,
                              void* d_out, int out_size, void* d_ws, size_t ws_size,
                              hipStream_t stream) {
  (void)in_sizes; (void)n_in; (void)out_size; (void)ws_size;
  const int*   x    = (const int*)d_in[0];
  const int*   ip   = (const int*)d_in[1];
  const float* emb  = (const float*)d_in[2];
  const float* pe   = (const float*)d_in[3];
  const float* Wq   = (const float*)d_in[4];
  const float* bq   = (const float*)d_in[5];
  const float* Wk   = (const float*)d_in[6];
  const float* bk   = (const float*)d_in[7];
  const float* Wv   = (const float*)d_in[8];
  const float* bv   = (const float*)d_in[9];
  const float* Wo   = (const float*)d_in[10];
  const float* bo   = (const float*)d_in[11];
  const float* ln1g = (const float*)d_in[12];
  const float* ln1b = (const float*)d_in[13];
  const float* l1w  = (const float*)d_in[14];
  const float* l1b  = (const float*)d_in[15];
  const float* l2w  = (const float*)d_in[16];
  const float* l2b  = (const float*)d_in[17];
  const float* ln2g = (const float*)d_in[18];
  const float* ln2b = (const float*)d_in[19];
  const float* lnfg = (const float*)d_in[20];
  const float* lnfb = (const float*)d_in[21];
  const float* pw   = (const float*)d_in[22];
  const float* pb   = (const float*)d_in[23];
  const float* Si   = (const float*)d_in[24];
  const float* Zi   = (const float*)d_in[25];

  // d_out: y_hat (512x768) | Si_out (6,512,8,64,64) | Zi_out (6,512,8,64)
  float* yhat = (float*)d_out;
  float* Sout = yhat + (size_t)512 * 768;
  float* Zout = Sout + (size_t)NLAYER * 512 * 8 * 64 * 64;

  // workspace (~10 MB), 256B aligned slabs
  char* p = (char*)d_ws;
  auto alloc = [&](size_t bytes) { char* r = p; p += (bytes + 255) & ~(size_t)255; return r; };
  float*    h    = (float*)alloc((size_t)512 * 512 * 4);
  float*    hn   = (float*)alloc((size_t)512 * 512 * 4);
  float*    Qf   = (float*)alloc((size_t)512 * 512 * 4);
  float*    Kf   = (float*)alloc((size_t)512 * 512 * 4);
  float*    Vf   = (float*)alloc((size_t)512 * 512 * 4);
  _Float16* h16  = (_Float16*)alloc((size_t)512 * 512 * 2);
  _Float16* a16  = (_Float16*)alloc((size_t)512 * 512 * 2);
  _Float16* y116 = (_Float16*)alloc((size_t)512 * 2048 * 2);
  _Float16* Wt   = (_Float16*)alloc((size_t)2048 * 512 * 2);  // reused slab

  auto conv = [&](const float* W, int K, int Nout) {
    int tot = K * Nout;
    k_convert_wt<<<(tot + 255) / 256, 256, 0, stream>>>(W, Wt, K, Nout);
  };
  auto gemm = [&](const _Float16* A, const float* bias, const float* resid,
                  float* Cf, _Float16* Ch, int M, int K, int Nout, int act) {
    k_gemm_wmma<<<dim3(M / 64, Nout / 64), 256, 0, stream>>>(
        A, Wt, bias, resid, Cf, Ch, M, K, Nout, act);
  };

  k_embed<<<(512 * 512) / 256, 256, 0, stream>>>(x, ip, emb, pe, h, h16);

  for (int l = 0; l < NLAYER; ++l) {
    const size_t wDD = (size_t)l * 512 * 512;
    conv(Wq + wDD, 512, 512);
    gemm(h16, bq + l * 512, nullptr, Qf, nullptr, 512, 512, 512, /*phi*/1);
    conv(Wk + wDD, 512, 512);
    gemm(h16, bk + l * 512, nullptr, Kf, nullptr, 512, 512, 512, /*phi*/1);
    conv(Wv + wDD, 512, 512);
    gemm(h16, bv + l * 512, nullptr, Vf, nullptr, 512, 512, 512, 0);

    const float* Sil = Si   + (size_t)l * 512 * 8 * 64 * 64;
    const float* Zil = Zi   + (size_t)l * 512 * 8 * 64;
    float*       Sol = Sout + (size_t)l * 512 * 8 * 64 * 64;
    float*       Zol = Zout + (size_t)l * 512 * 8 * 64;
    k_attn_state<<<512 * 8, 256, 0, stream>>>(Qf, Kf, Vf, Sil, Zil, Sol, Zol, a16);

    conv(Wo + wDD, 512, 512);
    gemm(a16, bo + l * 512, /*res=*/h, hn, nullptr, 512, 512, 512, 0);
    k_layernorm<<<512, 256, 0, stream>>>(hn, nullptr, ln1g + l * 512, ln1b + l * 512, h, h16);

    conv(l1w + (size_t)l * 512 * 2048, 512, 2048);
    gemm(h16, l1b + l * 2048, nullptr, nullptr, y116, 512, 512, 2048, /*relu*/2);
    conv(l2w + (size_t)l * 2048 * 512, 2048, 512);
    gemm(y116, l2b + l * 512, nullptr, hn, nullptr, 512, 2048, 512, 0);
    k_layernorm<<<512, 256, 0, stream>>>(hn, /*res=*/h, ln2g + l * 512, ln2b + l * 512, h, h16);
  }

  k_layernorm<<<512, 256, 0, stream>>>(h, nullptr, lnfg, lnfb, hn, h16);
  conv(pw, 512, 768);
  gemm(h16, pb, nullptr, yhat, nullptr, 512, 512, 768, 0);
}
